// FirstOrderCRF_84756884619424
// MI455X (gfx1250) — matrix operand order repaired and checked by reading, not compiled, not run on hardware
//
#include <hip/hip_runtime.h>
#include <hip/hip_bf16.h>
#include <math.h>

#define NUM_TAGS 128
#define BATCH    256
#define SEQ      1024

#define LOG2E 1.4426950408889634f
#define LN2   0.6931471805599453f

typedef __attribute__((ext_vector_type(16))) _Float16 v16h;
typedef __attribute__((ext_vector_type(8)))  float    v8f;

// Raw transcendentals: safe here because WMMA sums are >= e^-1 (no denormal
// inputs to log) and exp2 inputs are in [-~40, 0] (no overflow/denorm output).
__device__ __forceinline__ float fast_exp2(float x) { return __builtin_amdgcn_exp2f(x); }
__device__ __forceinline__ float fast_log2(float x) { return __builtin_amdgcn_logf(x); }

// ---------------------------------------------------------------------------
// Kernel 1: CRF forward (log-partition) via WMMA.
// Grid: BATCH/16 workgroups, 256 threads (8 wave32). Each workgroup owns a
// 16-row batch tile; each wave owns one 16-column tag tile (N-tile) and keeps
// E = exp(transitions) as 4 resident B-fragments (16x16x32 f16 WMMA).
// Per step:  alpha_new = emit_t + m + ln( exp(alpha - m) @ E )
// computed in log2 domain with raw v_exp_f32 / v_log_f32.
// ---------------------------------------------------------------------------
__global__ __launch_bounds__(256)
void crf_forward_kernel(const float* __restrict__ emissions,
                        const float* __restrict__ transitions,
                        const float* __restrict__ mask,
                        float* __restrict__ logZ)
{
    // a_lds: exp(alpha - m) as f16, [16 rows][128 tags], row stride 136 halves
    // (272 B) so the 16 rows land on distinct LDS banks for ds_load_b128.
    __shared__ __align__(16) _Float16 a_lds[16 * 136];
    __shared__ float mpart[8 * 16];   // per-wave row-max / row-sum partials
    __shared__ float mrow_lds[16];    // reduced per-row max

    const int tid   = threadIdx.x;
    const int wave  = tid >> 5;       // 0..7 == N tile
    const int lane  = tid & 31;
    const int lhalf = lane >> 4;      // 0: lanes 0-15, 1: lanes 16-31
    const int ln16  = lane & 15;
    const int b0    = blockIdx.x * 16;
    const int jcol  = (wave << 4) + ln16;   // this lane's tag column (B/C frag N)

    // ---- Precompute B fragments: E[k][j] = exp(T[k][j]), WMMA B layout ----
    // B (K=32 x N=16): lane holds column N=ln16; half h -> K = h + (h>=8?8:0) + 8*lhalf.
    v16h Bfrag[4];
#pragma unroll
    for (int kb = 0; kb < 4; ++kb) {
#pragma unroll
        for (int h = 0; h < 16; ++h) {
            const int k = kb * 32 + h + (h >= 8 ? 8 : 0) + lhalf * 8;
            Bfrag[kb][h] =
                (_Float16)fast_exp2(transitions[k * NUM_TAGS + jcol] * LOG2E);
        }
    }

    // ---- alpha(t=0) = emissions[:, 0, :]  (C-fragment layout) ----
    // C frag: VGPR r -> batch row = r + 8*lhalf, col = ln16 (-> jcol globally)
    float alpha[8];
#pragma unroll
    for (int r = 0; r < 8; ++r) {
        const int row = r + lhalf * 8;
        alpha[r] = emissions[(size_t)(b0 + row) * SEQ * NUM_TAGS + jcol];
    }

    for (int t = 1; t < SEQ; ++t) {
        // -- prefetch this step's emissions + mask (hidden under the barriers)
        float emit_r[8], mask_r[8];
#pragma unroll
        for (int r = 0; r < 8; ++r) {
            const int row = r + lhalf * 8;
            const size_t bt = (size_t)(b0 + row) * SEQ + t;
            emit_r[r] = emissions[bt * NUM_TAGS + jcol];
            mask_r[r] = mask[bt];
        }

        // -- per-wave row max over its 16 columns (butterfly within 16 lanes)
#pragma unroll
        for (int r = 0; r < 8; ++r) {
            float v = alpha[r];
            v = fmaxf(v, __shfl_xor(v, 1, 32));
            v = fmaxf(v, __shfl_xor(v, 2, 32));
            v = fmaxf(v, __shfl_xor(v, 4, 32));
            v = fmaxf(v, __shfl_xor(v, 8, 32));
            if (ln16 == r) mpart[wave * 16 + r + lhalf * 8] = v;
        }
        __syncthreads();                           // barrier 1

        // -- reduce row max across 8 waves
        if (tid < 16) {
            float mm = mpart[tid];
#pragma unroll
            for (int w = 1; w < 8; ++w) mm = fmaxf(mm, mpart[w * 16 + tid]);
            mrow_lds[tid] = mm;
        }
        __syncthreads();                           // barrier 2

        float mrows[8];
#pragma unroll
        for (int r = 0; r < 8; ++r) mrows[r] = mrow_lds[r + lhalf * 8];

        // -- a = exp(alpha - m) as f16 into LDS (each wave writes its tile)
#pragma unroll
        for (int r = 0; r < 8; ++r) {
            const int row = r + lhalf * 8;
            a_lds[row * 136 + jcol] =
                (_Float16)fast_exp2((alpha[r] - mrows[r]) * LOG2E);
        }
        __syncthreads();                           // barrier 3

        // -- S = A @ E : 4 chained v_wmma_f32_16x16x32_f16
        // A frag: M = ln16; half h -> K = 32*kb + h + (h>=8?8:0) + 8*lhalf,
        // i.e. two 16B-contiguous dword quads per kb -> 2x ds_load_b128.
        v8f c = {};
        const unsigned int* arow = (const unsigned int*)(a_lds + ln16 * 136);
#pragma unroll
        for (int kb = 0; kb < 4; ++kb) {
            union { v16h v; uint4 q[2]; } A;
            const int d0 = kb * 16 + lhalf * 4;    // dword index in row
            A.q[0] = *(const uint4*)(arow + d0);
            A.q[1] = *(const uint4*)(arow + d0 + 8);
            c = __builtin_amdgcn_wmma_f32_16x16x32_f16(
                    false, A.v, false, Bfrag[kb], (short)0, c, false, false);
        }

        // -- alpha_new = emit + m + ln(S); masked select against old alpha
#pragma unroll
        for (int r = 0; r < 8; ++r) {
            const float an = emit_r[r] + mrows[r] + fast_log2(c[r]) * LN2;
            alpha[r] = (mask_r[r] != 0.0f) ? an : alpha[r];
        }
        // Reuse of mpart/a_lds/mrow_lds next iteration is protected by
        // barriers 1-3 of that iteration.
    }

    // ---- logZ[row] = logsumexp_j alpha[row][j] ----
#pragma unroll
    for (int r = 0; r < 8; ++r) {
        float v = alpha[r];
        v = fmaxf(v, __shfl_xor(v, 1, 32));
        v = fmaxf(v, __shfl_xor(v, 2, 32));
        v = fmaxf(v, __shfl_xor(v, 4, 32));
        v = fmaxf(v, __shfl_xor(v, 8, 32));
        if (ln16 == r) mpart[wave * 16 + r + lhalf * 8] = v;
    }
    __syncthreads();
    if (tid < 16) {
        float mm = mpart[tid];
#pragma unroll
        for (int w = 1; w < 8; ++w) mm = fmaxf(mm, mpart[w * 16 + tid]);
        mrow_lds[tid] = mm;
    }
    __syncthreads();
    {
        float mrows[8];
#pragma unroll
        for (int r = 0; r < 8; ++r) mrows[r] = mrow_lds[r + lhalf * 8];
#pragma unroll
        for (int r = 0; r < 8; ++r) {
            float v = fast_exp2((alpha[r] - mrows[r]) * LOG2E);
            v += __shfl_xor(v, 1, 32);
            v += __shfl_xor(v, 2, 32);
            v += __shfl_xor(v, 4, 32);
            v += __shfl_xor(v, 8, 32);
            if (ln16 == r) mpart[wave * 16 + r + lhalf * 8] = v;  // row sums now
        }
    }
    __syncthreads();
    if (tid < 16) {
        float z = 0.0f;
#pragma unroll
        for (int w = 0; w < 8; ++w) z += mpart[w * 16 + tid];
        logZ[b0 + tid] = mrow_lds[tid] + fast_log2(z) * LN2;
    }
}

// ---------------------------------------------------------------------------
// Kernel 2: gold-path score per batch element (gather + reduce).
// ---------------------------------------------------------------------------
__global__ __launch_bounds__(256)
void crf_score_kernel(const float* __restrict__ emissions,
                      const float* __restrict__ transitions,
                      const int*   __restrict__ tags,
                      const float* __restrict__ mask,
                      float* __restrict__ score)
{
    __shared__ float red[256];
    const int b = blockIdx.x;
    const int tid = threadIdx.x;
    float acc = 0.0f;
    for (int t = tid; t < SEQ; t += 256) {
        const size_t bt = (size_t)b * SEQ + t;
        const int tg = tags[bt];
        const float mk = mask[bt];
        acc += emissions[bt * NUM_TAGS + tg] * mk;
        if (t >= 1) {
            const int tgp = tags[bt - 1];
            acc += transitions[tgp * NUM_TAGS + tg] * mk;
        }
    }
    red[tid] = acc;
    __syncthreads();
    for (int s = 128; s > 0; s >>= 1) {
        if (tid < s) red[tid] += red[tid + s];
        __syncthreads();
    }
    if (tid == 0) score[b] = red[0];
}

// ---------------------------------------------------------------------------
// Kernel 3: out = mean_b(logZ[b] - score[b])
// ---------------------------------------------------------------------------
__global__ __launch_bounds__(256)
void crf_final_kernel(const float* __restrict__ logZ,
                      const float* __restrict__ score,
                      float* __restrict__ out)
{
    __shared__ float red[256];
    const int tid = threadIdx.x;
    red[tid] = logZ[tid] - score[tid];
    __syncthreads();
    for (int s = 128; s > 0; s >>= 1) {
        if (tid < s) red[tid] += red[tid + s];
        __syncthreads();
    }
    if (tid == 0) out[0] = red[0] * (1.0f / (float)BATCH);
}

extern "C" void kernel_launch(void* const* d_in, const int* in_sizes, int n_in,
                              void* d_out, int out_size, void* d_ws, size_t ws_size,
                              hipStream_t stream)
{
    const float* emissions   = (const float*)d_in[0];
    const float* transitions = (const float*)d_in[1];
    const int*   tags        = (const int*)d_in[2];
    const float* mask        = (const float*)d_in[3];
    float* out = (float*)d_out;

    float* ws    = (float*)d_ws;
    float* logZ  = ws;            // [BATCH]
    float* score = ws + BATCH;    // [BATCH]

    crf_forward_kernel<<<BATCH / 16, 256, 0, stream>>>(emissions, transitions, mask, logZ);
    crf_score_kernel<<<BATCH, 256, 0, stream>>>(emissions, transitions, tags, mask, score);
    crf_final_kernel<<<1, 256, 0, stream>>>(logZ, score, out);
}
